// Decoder_20435454395154
// MI455X (gfx1250) — matrix operand order repaired and checked by reading, not compiled, not run on hardware
//
#include <hip/hip_runtime.h>
#include <hip/hip_bf16.h>
#include <math.h>

// ---------------- CDNA5 WMMA plumbing (wave32, bf16 16x16x32, f32 accum) ----
typedef __attribute__((ext_vector_type(16))) __bf16 v16bf;
typedef __attribute__((ext_vector_type(16))) float  v16f;
typedef __attribute__((ext_vector_type(8)))  float  v8f;

struct BF16x16 { uint4 lo, hi; };   // 32 bytes = 16 bf16

// Native bf16 convert (RNE fptrunc -> v_cvt_pk_bf16_f32 on gfx1250)
__device__ __forceinline__ unsigned short f2bf(float f) {
    __bf16 h = (__bf16)f;
    return __builtin_bit_cast(unsigned short, h);
}
__device__ __forceinline__ float bf2f(unsigned short h) {
    return __uint_as_float(((unsigned)h) << 16);
}

__device__ __forceinline__ v8f wmma_bf16(v16bf a, v16bf b, v8f c) {
    // v_wmma_f32_16x16x32_bf16  (neg_a, A, neg_b, B, c_mod, C, reuse_a, reuse_b)
    return __builtin_amdgcn_wmma_f32_16x16x32_bf16(false, a, false, b, (short)0, c,
                                                   false, false);
}

// A-matrix 16x32 bf16 fragment: lane<16 -> row M=lane, K=k0+0..15
//                               lane>=16 -> row M=lane-16, K=k0+16..31
__device__ __forceinline__ v16bf load_A(const unsigned short* base, int row0, int ld,
                                        int k0, int lane) {
    int l = lane & 15, hk = lane >> 4;
    const unsigned short* p = base + (size_t)(row0 + l) * ld + k0 + 16 * hk;
    BF16x16 r;
    r.lo = *(const uint4*)p;
    r.hi = *(const uint4*)(p + 8);
    return __builtin_bit_cast(v16bf, r);
}

// B-matrix 32x16 bf16 fragment from row-major W[n][k]:
// lane<16 -> col N=lane, K=k0+0..15 ; lane>=16 -> col N=lane-16, K=k0+16..31
__device__ __forceinline__ v16bf load_B(const unsigned short* base, int n0, int ld,
                                        int k0, int lane) {
    int l = lane & 15, hk = lane >> 4;
    const unsigned short* p = base + (size_t)(n0 + l) * ld + k0 + 16 * hk;
    BF16x16 r;
    r.lo = *(const uint4*)p;
    r.hi = *(const uint4*)(p + 8);
    return __builtin_bit_cast(v16bf, r);
}

// Same B fragment but converting f32 -> bf16 on the fly (single-use weights).
// 64B vector load + packed convert (v_cvt_pk_bf16_f32).
__device__ __forceinline__ v16bf load_B_f32(const float* base, int n0, int ld,
                                            int k0, int lane) {
    int l = lane & 15, hk = lane >> 4;
    const float* p = base + (size_t)(n0 + l) * ld + k0 + 16 * hk;  // 64B aligned
    v16f t = *(const v16f*)p;
    return __builtin_convertvector(t, v16bf);
}

__device__ __forceinline__ float sigm(float x) { return 1.f / (1.f + __expf(-x)); }

// ---------------- Problem constants ----------------------------------------
#define BATCH   32
#define SEQ     3072
#define DIN     64
#define DH      256
#define NGATE   1024      // 4*DH
#define NDEC    65536     // 64*32*32
#define NVOC    256
#define NWG_SCAN 16

// ---------------- K0: weight-norm + bf16 weight prep + barrier reset --------
__global__ void k_prep(const float* v_ih, const float* g_ih,
                       const float* v_hh, const float* g_hh,
                       const float* b_ih, const float* b_hh,
                       const float* W_ro,
                       unsigned short* Wih_bf, unsigned short* Whh_bf,
                       unsigned short* Wro_bf, float* bias, unsigned* bar) {
    int r = blockIdx.x * blockDim.x + threadIdx.x;   // 0..1023
    if (r < 2) bar[r] = 0u;                          // reset spin barrier each launch
    if (r >= NGATE) return;
    // W_ih row (64 wide)
    float s = 0.f;
    for (int k = 0; k < DIN; ++k) { float v = v_ih[r * DIN + k]; s += v * v; }
    float inv = g_ih[r] * rsqrtf(s);
    for (int k = 0; k < DIN; ++k) Wih_bf[r * DIN + k] = f2bf(v_ih[r * DIN + k] * inv);
    // W_hh row (256 wide)
    s = 0.f;
    for (int k = 0; k < DH; ++k) { float v = v_hh[r * DH + k]; s += v * v; }
    inv = g_hh[r] * rsqrtf(s);
    for (int k = 0; k < DH; ++k) Whh_bf[r * DH + k] = f2bf(v_hh[r * DH + k] * inv);
    bias[r] = b_ih[r] + b_hh[r];
    // readout weights (no norm), bf16
    if (r < NVOC)
        for (int k = 0; k < DH; ++k) Wro_bf[r * DH + k] = f2bf(W_ro[r * DH + k]);
}

// ---------------- K1: conditioning  cond = z + theta@W_theta.T + b ----------
__global__ void k_cond(const float* z, const float* theta, const float* W_theta,
                       const float* b_theta, unsigned short* cond_bf) {
    int idx = blockIdx.x * blockDim.x + threadIdx.x;       // 32*256
    if (idx >= BATCH * 256) return;
    int b = idx >> 8, k = idx & 255;
    float v = z[idx] + theta[b * 2] * W_theta[k * 2] +
              theta[b * 2 + 1] * W_theta[k * 2 + 1] + b_theta[k];
    cond_bf[idx] = f2bf(v);
}

// ---------------- K2: decoder GEMM  x[32,65536] = cond @ W_dec.T + b_dec ----
__global__ void k_dec(const unsigned short* cond_bf, const float* W_dec,
                      const float* b_dec, float* xbuf) {
    int tid = threadIdx.x, lane = tid & 31;
    int wid = blockIdx.x * 8 + (tid >> 5);                 // 4096 N-tiles total
    int n0 = wid * 16;
    v8f c0 = {}, c1 = {};
    for (int k0 = 0; k0 < 256; k0 += 32) {
        v16bf a0 = load_A(cond_bf, 0, 256, k0, lane);
        v16bf a1 = load_A(cond_bf, 16, 256, k0, lane);
        __builtin_prefetch(W_dec + (size_t)(n0 + (lane & 15)) * 256 + k0 + 32, 0, 1);
        v16bf bf = load_B_f32(W_dec, n0, 256, k0, lane);
        c0 = wmma_bf16(a0, bf, c0);
        c1 = wmma_bf16(a1, bf, c1);
    }
    int l = lane & 15, hk = lane >> 4;
    int n = n0 + l;
    float bd = b_dec[n];
#pragma unroll
    for (int r = 0; r < 8; ++r) {
        xbuf[(size_t)(r + 8 * hk) * NDEC + n]      = c0[r] + bd;
        xbuf[(size_t)(16 + r + 8 * hk) * NDEC + n] = c1[r] + bd;
    }
}

// ---------------- K3: GroupNorm(8,64) statistics ----------------------------
__global__ void k_gnstats(const float* xbuf, float* mu, float* rstd) {
    __shared__ float s1[256], s2[256];
    int bg = blockIdx.x;                       // b*8+g, 256 blocks
    const float* p = xbuf + (size_t)(bg >> 3) * NDEC + (size_t)(bg & 7) * 8192;
    float a = 0.f, q = 0.f;
    for (int i = threadIdx.x; i < 8192; i += 256) { float v = p[i]; a += v; q += v * v; }
    s1[threadIdx.x] = a; s2[threadIdx.x] = q;
    __syncthreads();
    for (int s = 128; s > 0; s >>= 1) {
        if (threadIdx.x < s) { s1[threadIdx.x] += s1[threadIdx.x + s];
                               s2[threadIdx.x] += s2[threadIdx.x + s]; }
        __syncthreads();
    }
    if (threadIdx.x == 0) {
        float m = s1[0] * (1.f / 8192.f);
        float var = s2[0] * (1.f / 8192.f) - m * m;
        mu[bg] = m;
        rstd[bg] = rsqrtf(var + 1e-5f);
    }
}

// ---------------- K4: build LSTM input (norm'd features + shifted emb) ------
__global__ void k_inp(const float* xbuf, const float* mu, const float* rstd,
                      const float* gn_w, const float* gn_b, const float* emb,
                      const int* x_target, unsigned short* inp_bf) {
    size_t idx = (size_t)blockIdx.x * blockDim.x + threadIdx.x;  // 32*3072*64
    if (idx >= (size_t)BATCH * SEQ * DIN) return;
    int d = idx & 63;
    int t = (int)((idx >> 6) % SEQ);
    int b = (int)(idx / ((size_t)SEQ * DIN));
    int pix = t & 1023;                                  // t % (32*32)
    int g = b * 8 + (d >> 3);
    float xn = (xbuf[(size_t)b * NDEC + d * 1024 + pix] - mu[g]) * rstd[g] * gn_w[d] +
               gn_b[d];
    float e = 0.f;
    if (t > 0) e = emb[(size_t)x_target[b * SEQ + t - 1] * DIN + d];
    inp_bf[idx] = f2bf(xn + e);
}

// ---------------- K5: pre = inp @ W_ih.T + bias  (stored [t][n][b] bf16) ----
__global__ void k_pre(const unsigned short* inp_bf, const unsigned short* Wih_bf,
                      const float* bias, unsigned short* pre_T) {
    int tid = threadIdx.x, lane = tid & 31;
    int wid = blockIdx.x * 8 + (tid >> 5);      // 6144 mtiles * 64 ntiles waves
    int mt = wid >> 6, nt = wid & 63;
    int row0 = mt * 16, n0 = nt * 16;
    int l = lane & 15, hk = lane >> 4;
    float bv = bias[n0 + l];
    v8f c = {};
#pragma unroll
    for (int r = 0; r < 8; ++r) c[r] = bv;
    for (int k0 = 0; k0 < DIN; k0 += 32) {
        v16bf a  = load_A(inp_bf, row0, DIN, k0, lane);
        v16bf bf = load_B(Wih_bf, n0, DIN, k0, lane);
        c = wmma_bf16(a, bf, c);
    }
    int bb = row0 / SEQ;
    int tbase = row0 % SEQ + 8 * hk;
#pragma unroll
    for (int r = 0; r < 8; ++r) {
        int tt = tbase + r;
        pre_T[((size_t)tt * NGATE + n0 + l) * BATCH + bb] = f2bf(c[r]);
    }
}

// ---------------- scan: cross-WGP step barrier ------------------------------
__device__ __forceinline__ void gsync(volatile unsigned* bar, unsigned gen) {
    __syncthreads();
    if (threadIdx.x == 0) {
        __threadfence();                                   // release h/hs stores
        unsigned old = atomicAdd((unsigned*)&bar[0], 1u);
        if (old == NWG_SCAN - 1) {
            bar[0] = 0u;
            __threadfence();
            atomicExch((unsigned*)&bar[1], gen);
        } else {
            while (bar[1] < gen) __builtin_amdgcn_s_sleep(1);
        }
    }
    __builtin_amdgcn_s_cluster_barrier();   // s_barrier_signal/-wait -3 (NOP if unclustered)
    __syncthreads();
    __threadfence();                                       // acquire (invalidate WGP$)
}

// ---------------- K6: sequential LSTM scan, 16 WGs, WMMA recurrence ---------
__global__ void k_scan(const unsigned short* pre_T, const unsigned short* Whh_bf,
                       unsigned short* hbuf, unsigned short* hs_bf, unsigned* bar) {
    __shared__ unsigned short Whh_s[64][264];   // 64 gate rows x 256 K (+8 pad)
    __shared__ float gates_s[4][32][16];
    __shared__ float c_s[32][16];

    const int w = blockIdx.x;                   // 0..15 : owns h cols [16w,16w+16)
    const int tid = threadIdx.x, lane = tid & 31, wv = tid >> 5;
    const int m = wv & 1, gi = wv >> 1;         // wave -> (M-tile, gate)
    const int l = lane & 15, hk = lane >> 4;

    // stage this WG's W_hh slice into LDS (resident for all 3072 steps)
    for (int i = tid; i < 64 * 256; i += 256) {
        int gc = i >> 8, k = i & 255;
        int grow = (gc >> 4) * DH + w * 16 + (gc & 15);    // gate*(256) + col
        Whh_s[gc][k] = Whh_bf[grow * DH + k];
    }
    for (int i = tid; i < 512; i += 256) {
        c_s[i >> 4][i & 15] = 0.f;
        hbuf[(i >> 4) * DH + w * 16 + (i & 15)] = 0;       // h0 = 0 (buffer 0)
    }
    gsync(bar, 1u);

    const int ng0 = gi * DH + w * 16;                      // global gate-col base
    for (int t = 0; t < SEQ; ++t) {
        const unsigned short* hread  = hbuf + (size_t)(t & 1) * BATCH * DH;
        unsigned short*       hwrite = hbuf + (size_t)((t + 1) & 1) * BATCH * DH;

        // accumulator init from precomputed input projection (vector load)
        const unsigned short* pp =
            pre_T + ((size_t)t * NGATE + ng0 + l) * BATCH + m * 16 + 8 * hk;
        uint4 pv = *(const uint4*)pp;
        unsigned pw[4] = {pv.x, pv.y, pv.z, pv.w};
        v8f c = {};
#pragma unroll
        for (int r = 0; r < 8; ++r)
            c[r] = bf2f((unsigned short)((pw[r >> 1] >> ((r & 1) * 16)) & 0xFFFFu));

        // gates += h @ W_hh.T  : 8 chained v_wmma_f32_16x16x32_bf16
#pragma unroll
        for (int k0 = 0; k0 < DH; k0 += 32) {
            v16bf a  = load_A(hread, m * 16, DH, k0, lane);
            v16bf bf = load_B(&Whh_s[0][0], gi * 16, 264, k0, lane);
            c = wmma_bf16(a, bf, c);
        }
#pragma unroll
        for (int r = 0; r < 8; ++r) gates_s[gi][m * 16 + 8 * hk + r][l] = c[r];
        __syncthreads();

        // LSTM cell (i,f,g,o) for this WG's 32x16 slice
        for (int e = tid; e < 512; e += 256) {
            int b = e >> 4, col = e & 15;
            float iv = gates_s[0][b][col], fv = gates_s[1][b][col];
            float gv = gates_s[2][b][col], ov = gates_s[3][b][col];
            float cn = sigm(fv) * c_s[b][col] + sigm(iv) * tanhf(gv);
            float hv = sigm(ov) * tanhf(cn);
            c_s[b][col] = cn;
            unsigned short hb = f2bf(hv);
            int colg = w * 16 + col;
            hwrite[b * DH + colg] = hb;
            hs_bf[((size_t)b * SEQ + t) * DH + colg] = hb;
        }
        gsync(bar, (unsigned)(t + 2));
    }
}

// ---------------- K7: out = hs @ W_ro.T + b_ro, fused [b,v,t] permute -------
__global__ void k_out(const unsigned short* hs_bf, const unsigned short* Wro_bf,
                      const float* b_ro, float* out) {
    int tid = threadIdx.x, lane = tid & 31;
    int wid = blockIdx.x * 8 + (tid >> 5);       // 6144 mtiles * 16 ntiles
    int mt = wid >> 4, nt = wid & 15;
    int row0 = mt * 16, n0 = nt * 16;
    int l = lane & 15, hk = lane >> 4;
    float bv = b_ro[n0 + l];
    v8f c = {};
#pragma unroll
    for (int r = 0; r < 8; ++r) c[r] = bv;
    for (int k0 = 0; k0 < DH; k0 += 32) {
        v16bf a  = load_A(hs_bf, row0, DH, k0, lane);
        v16bf bf = load_B(Wro_bf, n0, DH, k0, lane);
        c = wmma_bf16(a, bf, c);
    }
    int bb = row0 / SEQ;
    int tbase = row0 % SEQ + 8 * hk;
    size_t base = ((size_t)bb * NVOC + n0 + l) * SEQ + tbase;
#pragma unroll
    for (int r = 0; r < 8; ++r) out[base + r] = c[r];     // [b][v][c3*1024+pix]
}

// ---------------- host: workspace carve + launches --------------------------
static inline size_t align256(size_t x) { return (x + 255) & ~(size_t)255; }

extern "C" void kernel_launch(void* const* d_in, const int* in_sizes, int n_in,
                              void* d_out, int out_size, void* d_ws, size_t ws_size,
                              hipStream_t stream) {
    const float* z_noisy  = (const float*)d_in[0];
    const float* theta    = (const float*)d_in[1];
    const float* W_theta  = (const float*)d_in[2];
    const float* b_theta  = (const float*)d_in[3];
    const float* W_dec    = (const float*)d_in[4];
    const float* b_dec    = (const float*)d_in[5];
    const float* gn_w     = (const float*)d_in[6];
    const float* gn_b     = (const float*)d_in[7];
    const float* emb      = (const float*)d_in[8];
    const float* v_ih     = (const float*)d_in[9];
    const float* g_ih     = (const float*)d_in[10];
    const float* v_hh     = (const float*)d_in[11];
    const float* g_hh     = (const float*)d_in[12];
    const float* b_ih     = (const float*)d_in[13];
    const float* b_hh     = (const float*)d_in[14];
    const float* W_ro     = (const float*)d_in[15];
    const float* b_ro     = (const float*)d_in[16];
    const int*   x_target = (const int*)d_in[17];
    float* out = (float*)d_out;

    char* p = (char*)d_ws;
    size_t off = 0;
    auto carve = [&](size_t bytes) { void* r = p + off; off = align256(off + bytes); return r; };
    unsigned short* cond_bf = (unsigned short*)carve((size_t)BATCH * 256 * 2);
    unsigned short* Wih_bf  = (unsigned short*)carve((size_t)NGATE * DIN * 2);
    unsigned short* Whh_bf  = (unsigned short*)carve((size_t)NGATE * DH * 2);
    unsigned short* Wro_bf  = (unsigned short*)carve((size_t)NVOC * DH * 2);
    float*          bias    = (float*)carve((size_t)NGATE * 4);
    float*          xbuf    = (float*)carve((size_t)BATCH * NDEC * 4);
    float*          mu      = (float*)carve(256 * 4);
    float*          rstd    = (float*)carve(256 * 4);
    unsigned short* inp_bf  = (unsigned short*)carve((size_t)BATCH * SEQ * DIN * 2);
    unsigned short* pre_T   = (unsigned short*)carve((size_t)SEQ * NGATE * BATCH * 2);
    unsigned short* hbuf    = (unsigned short*)carve((size_t)2 * BATCH * DH * 2);
    unsigned short* hs_bf   = (unsigned short*)carve((size_t)BATCH * SEQ * DH * 2);
    unsigned*       bar     = (unsigned*)carve(256);
    (void)ws_size; (void)in_sizes; (void)n_in; (void)out_size;

    k_prep<<<4, 256, 0, stream>>>(v_ih, g_ih, v_hh, g_hh, b_ih, b_hh, W_ro,
                                  Wih_bf, Whh_bf, Wro_bf, bias, bar);
    k_cond<<<32, 256, 0, stream>>>(z_noisy, theta, W_theta, b_theta, cond_bf);
    k_dec<<<512, 256, 0, stream>>>(cond_bf, W_dec, b_dec, xbuf);
    k_gnstats<<<256, 256, 0, stream>>>(xbuf, mu, rstd);
    k_inp<<<(BATCH * SEQ * DIN + 255) / 256, 256, 0, stream>>>(
        xbuf, mu, rstd, gn_w, gn_b, emb, x_target, inp_bf);
    k_pre<<<(6144 * 64) / 8, 256, 0, stream>>>(inp_bf, Wih_bf, bias, pre_T);
    k_scan<<<NWG_SCAN, 256, 0, stream>>>(pre_T, Whh_bf, hbuf, hs_bf, bar);
    k_out<<<(6144 * 16) / 8, 256, 0, stream>>>(hs_bf, Wro_bf, b_ro, out);
}
